// OftNet_79937931313291
// MI455X (gfx1250) — matrix operand order repaired and checked by reading, not compiled, not run on hardware
//
#include <hip/hip_runtime.h>

typedef __attribute__((ext_vector_type(16))) _Float16 v16h;
typedef __attribute__((ext_vector_type(8)))  _Float16 v8h;
typedef __attribute__((ext_vector_type(8)))  float    v8f;

#define NLOC  13392      // D*Wd = 108*124
#define ZM    9
#define DD    108
#define WWD   124
#define NCH   1280       // 5 levels * 256 channels
#define KORT  2304       // 256*9
#define MPORT 13440      // NLOC padded to 128
#define MCONV 3348       // 62*54
#define MPCON 3456       // padded to 128
#define K1    11520      // 1280*9
#define K2    2304       // 256*9
#define OH    62
#define OW    54
#define NGRP  64         // partial-sum groups for channel pooling
#define MCHUNK 210       // ceil(NLOC/NGRP)

// ---------------------------------------------------------------- utilities
__global__ void zero_f16_kernel(_Float16* __restrict__ p, int n) {
    int i = blockIdx.x * blockDim.x + threadIdx.x;
    if (i < n) p[i] = (_Float16)0.0f;
}

__global__ void cvt_f16_kernel(const float* __restrict__ s, _Float16* __restrict__ d, int n) {
    int i = blockIdx.x * blockDim.x + threadIdx.x;
    if (i < n) d[i] = (_Float16)s[i];
}

// ------------------------------------------------- integral image (H,W,C layout)
// pass 1: cumsum along W, transposing (C,H,W) -> (H,W,C)
__global__ void cumsum_w_kernel(const float* __restrict__ f, float* __restrict__ ii,
                                int H, int W) {
    int h = blockIdx.x;
    int c = threadIdx.x;                 // 256 channels
    const float* src = f + (size_t)c * H * W + (size_t)h * W;
    float run = 0.0f;
    for (int w = 0; w < W; ++w) {
        run += src[w];
        ii[((size_t)h * W + w) * 256 + c] = run;
    }
}

// pass 2: cumsum along H, in place on (H,W,C)
__global__ void cumsum_h_kernel(float* __restrict__ ii, int H, int W) {
    int w = blockIdx.x;
    int c = threadIdx.x;
    size_t stride = (size_t)W * 256;
    float* p = ii + (size_t)w * 256 + c;
    float run = p[0];
    for (int h = 1; h < H; ++h) {
        run += p[(size_t)h * stride];
        p[(size_t)h * stride] = run;
    }
}

// ------------------------------------------------- bbox from calib (scale-invariant)
__device__ __forceinline__ void project_norm(const float* __restrict__ c,
                                             int zi, int xi, int yi,
                                             float& nx, float& ny) {
    float X = xi * 0.64f;
    float Y = 39.68f - yi * 0.64f;
    float Z = 2.76f - zi * 0.64f;
    float hx = c[0] * X + c[1] * Y + c[2]  * Z + c[3];
    float hy = c[4] * X + c[5] * Y + c[6]  * Z + c[7];
    float hz = c[8] * X + c[9] * Y + c[10] * Z + c[11];
    float ix = hx / hz, iy = hy / hz;
    // img_size = [W/scale, H/scale] = [1280, 384] for every pyramid level
    nx = fminf(fmaxf(ix * (2.0f / 1280.0f) - 1.0f, -1.0f), 1.0f);
    ny = fminf(fmaxf(iy * (2.0f / 384.0f)  - 1.0f, -1.0f), 1.0f);
}

__global__ void bbox_kernel(const float* __restrict__ calib, float* __restrict__ bbox) {
    int idx = blockIdx.x * blockDim.x + threadIdx.x;
    if (idx >= ZM * NLOC) return;
    int z   = idx / NLOC;
    int loc = idx % NLOC;
    int d   = loc / WWD;
    int wd  = loc % WWD;
    float c[12];
#pragma unroll
    for (int i = 0; i < 12; ++i) c[i] = calib[i];
    float ax, ay, bx, by, cx, cy, dx, dy;
    project_norm(c, z,     d,     wd,     ax, ay);   // norm[z, d,   wd]
    project_norm(c, z,     d + 1, wd,     bx, by);   // norm[z, d+1, wd]
    project_norm(c, z + 1, d + 1, wd + 1, cx, cy);   // norm[z+1, d+1, wd+1]
    project_norm(c, z + 1, d,     wd + 1, dx, dy);   // norm[z+1, d,   wd+1]
    bbox[(size_t)idx * 4 + 0] = fminf(ax, bx);
    bbox[(size_t)idx * 4 + 1] = fminf(ay, by);
    bbox[(size_t)idx * 4 + 2] = fmaxf(cx, dx);
    bbox[(size_t)idx * 4 + 3] = fmaxf(cy, dy);
}

// ------------------------------------------------- voxel sampling -> f16 A matrix
__device__ __forceinline__ float tap(const float* __restrict__ ii,
                                     int xi, int yi, int W, int H, int c) {
    bool ok = (xi >= 0) && (xi < W) && (yi >= 0) && (yi < H);
    int xc = min(max(xi, 0), W - 1);
    int yc = min(max(yi, 0), H - 1);
    float v = ii[((size_t)yc * W + xc) * 256 + c];
    return ok ? v : 0.0f;
}

__device__ __forceinline__ float bilin(const float* __restrict__ ii,
                                       float gx, float gy, int W, int H, int c) {
    float x  = ((gx + 1.0f) * W - 1.0f) * 0.5f;
    float y  = ((gy + 1.0f) * H - 1.0f) * 0.5f;
    float x0 = floorf(x), y0 = floorf(y);
    float wx = x - x0,    wy = y - y0;
    int xi = (int)x0, yi = (int)y0;
    float v00 = tap(ii, xi,     yi,     W, H, c);
    float v10 = tap(ii, xi + 1, yi,     W, H, c);
    float v01 = tap(ii, xi,     yi + 1, W, H, c);
    float v11 = tap(ii, xi + 1, yi + 1, W, H, c);
    return v00 * (1.0f - wx) * (1.0f - wy) + v10 * wx * (1.0f - wy)
         + v01 * (1.0f - wx) * wy          + v11 * wx * wy;
}

__global__ __launch_bounds__(256)
void vox_kernel(const float* __restrict__ ii, const float* __restrict__ bbox,
                _Float16* __restrict__ A, int H, int W) {
    int b   = blockIdx.x;                 // z * NLOC + loc
    int z   = b / NLOC;
    int loc = b % NLOC;
    const float* bb = bbox + (size_t)b * 4;
    float x1 = bb[0], y1 = bb[1], x2 = bb[2], y2 = bb[3];
    float area = (x2 - x1) * (y2 - y1) * (float)(H * W) * 0.25f + 1e-6f;
    float vis  = (area > 1e-6f) ? 1.0f : 0.0f;
    float inv  = vis / area;
    int c = threadIdx.x;                  // channel
    float tl = bilin(ii, x1, y1, W, H, c);
    float br = bilin(ii, x2, y2, W, H, c);
    float tr = bilin(ii, x2, y1, W, H, c);
    float bl = bilin(ii, x1, y2, W, H, c);
    float v  = (tl + br - tr - bl) * inv;
    A[(size_t)loc * KORT + c * ZM + z] = (_Float16)v;   // k = c*9 + z
}

// ------------------------------------------------- WMMA GEMM: out = A(MxK,f16) * B(N-major K,f16)
// mode 0: ortho  -> relu(acc + p0[n]) stored m-major: out[m*1280 + n]   (coalesced)
// mode 1: bn+relu-> relu(acc*sc + bi) stored n-major: out[n*Mreal + m]
__global__ __launch_bounds__(256)
void gemm_wmma_kernel(const _Float16* __restrict__ A, const _Float16* __restrict__ B,
                      int Mreal, int K, int mode,
                      const float* __restrict__ p0, const float* __restrict__ p1,
                      const float* __restrict__ p2, const float* __restrict__ p3,
                      float* __restrict__ out) {
    const int lane  = threadIdx.x & 31;
    const int wave  = threadIdx.x >> 5;
    const int mbase = blockIdx.x * 128 + wave * 16;
    const int nbase = blockIdx.y * 64;
    const int khalf = lane >> 4;          // which K-half this lane supplies
    const int l16   = lane & 15;

    const _Float16* Arow = A + (size_t)(mbase + l16) * K;
    const _Float16* Br0  = B + (size_t)(nbase +  0 + l16) * K;
    const _Float16* Br1  = B + (size_t)(nbase + 16 + l16) * K;
    const _Float16* Br2  = B + (size_t)(nbase + 32 + l16) * K;
    const _Float16* Br3  = B + (size_t)(nbase + 48 + l16) * K;

    v8f acc0 = {}, acc1 = {}, acc2 = {}, acc3 = {};

    for (int kb = 0; kb < K; kb += 32) {
        __builtin_prefetch(Arow + kb + 8 * khalf + 512, 0, 3);   // global_prefetch_b8 (near)
        // A 16x32 f16 fragment: two contiguous 8-halve chunks per lane
        v8h alo = *(const v8h*)(Arow + kb + 16 * khalf);
        v8h ahi = *(const v8h*)(Arow + kb + 16 * khalf + 8);
        v16h av = __builtin_shufflevector(alo, ahi,
                    0, 1, 2, 3, 4, 5, 6, 7, 8, 9, 10, 11, 12, 13, 14, 15);
        // B 32x16 fragments: one contiguous 16-halve chunk per lane (N-major storage)
        int kof = kb + 16 * khalf;
        v16h b0 = *(const v16h*)(Br0 + kof);
        v16h b1 = *(const v16h*)(Br1 + kof);
        v16h b2 = *(const v16h*)(Br2 + kof);
        v16h b3 = *(const v16h*)(Br3 + kof);
        acc0 = __builtin_amdgcn_wmma_f32_16x16x32_f16(false, av, false, b0, (short)0, acc0, false, false);
        acc1 = __builtin_amdgcn_wmma_f32_16x16x32_f16(false, av, false, b1, (short)0, acc1, false, false);
        acc2 = __builtin_amdgcn_wmma_f32_16x16x32_f16(false, av, false, b2, (short)0, acc2, false, false);
        acc3 = __builtin_amdgcn_wmma_f32_16x16x32_f16(false, av, false, b3, (short)0, acc3, false, false);
    }

    v8f accs[4] = {acc0, acc1, acc2, acc3};
    const int mrow0 = mbase + khalf * 8;            // C layout: m = r + 8*(lane/16)
#pragma unroll
    for (int j = 0; j < 4; ++j) {
        int n = nbase + j * 16 + l16;               // n = lane%16
        if (mode == 0) {
            float bias = p0[n];
#pragma unroll
            for (int r = 0; r < 8; ++r) {
                int m = mrow0 + r;
                float v = accs[j][r] + bias;
                v = v > 0.0f ? v : 0.0f;
                if (m < Mreal) out[(size_t)m * NCH + n] = v;   // m-major, coalesced
            }
        } else {
            float sc = p0[n] * rsqrtf(p3[n] + 1e-5f);
            float bi = p1[n] - p2[n] * sc;
#pragma unroll
            for (int r = 0; r < 8; ++r) {
                int m = mrow0 + r;
                float v = accs[j][r] * sc + bi;
                v = v > 0.0f ? v : 0.0f;
                if (m < Mreal) out[(size_t)n * Mreal + m] = v;
            }
        }
    }
}

// ------------------------------------------------- deterministic coalesced channel pooling
// pass 1: 64 groups of rows, thread t owns channels {t, 256+t, ..., 1024+t}
__global__ void chansum_part_kernel(const float* __restrict__ x, float* __restrict__ part) {
    int g = blockIdx.x;
    int t = threadIdx.x;
    float acc[5] = {0.f, 0.f, 0.f, 0.f, 0.f};
    int m0 = g * MCHUNK;
    int m1 = min(m0 + MCHUNK, NLOC);
    for (int m = m0; m < m1; ++m) {
        const float* row = x + (size_t)m * NCH;
#pragma unroll
        for (int s = 0; s < 5; ++s) acc[s] += row[s * 256 + t];
    }
#pragma unroll
    for (int s = 0; s < 5; ++s) part[(size_t)g * NCH + s * 256 + t] = acc[s];
}

__global__ void chansum_final_kernel(const float* __restrict__ part, float* __restrict__ cs) {
    int ch = blockIdx.x * 256 + threadIdx.x;     // 5 blocks x 256
    float s = 0.0f;
    for (int g = 0; g < NGRP; ++g) s += part[(size_t)g * NCH + ch];
    cs[ch] = s;
}

__global__ void att_kernel(const float* __restrict__ cs, const float* __restrict__ att_w,
                           const float* __restrict__ att_b, float* __restrict__ attv) {
    int s = blockIdx.x;                   // pyramid level
    int i = threadIdx.x;                  // out channel
    const float* wrow = att_w + (size_t)i * 256;
    const float* ps   = cs + (size_t)s * 256;
    float acc = att_b[i];
    for (int j = 0; j < 256; ++j) acc += wrow[j] * (ps[j] * (1.0f / (float)NLOC));
    attv[s * 256 + i] = 1.0f / (1.0f + expf(-acc));
}

// ------------------------------------------------- im2col (attention folded into conv1 input)
// ortho is m-major: x[(d*124 + wd) * 1280 + ch]; spatial map: iy=wd (H'=124), ix=d (W'=108)
__global__ void im2col1_kernel(const float* __restrict__ x, const float* __restrict__ attv,
                               _Float16* __restrict__ A1) {
    size_t idx = (size_t)blockIdx.x * 256 + threadIdx.x;   // MPCON * K1 elements
    int m = (int)(idx / K1);
    int k = (int)(idx % K1);
    int cin = k / 9, t = k % 9, ky = t / 3, kx = t % 3;
    int oy = m / OW, ox = m % OW;
    int iy = oy * 2 + ky - 1;
    int ix = ox * 2 + kx - 1;
    float v = 0.0f;
    if (m < MCONV && iy >= 0 && iy < WWD && ix >= 0 && ix < DD)
        v = x[((size_t)ix * WWD + iy) * NCH + cin] * attv[cin];
    A1[idx] = (_Float16)v;
}

__global__ void im2col2_kernel(const float* __restrict__ y1, _Float16* __restrict__ A2) {
    size_t idx = (size_t)blockIdx.x * 256 + threadIdx.x;   // MPCON * K2 elements
    int m = (int)(idx / K2);
    int k = (int)(idx % K2);
    int cin = k / 9, t = k % 9, ky = t / 3, kx = t % 3;
    int oy = m / OW, ox = m % OW;
    int iy = oy + ky - 1;
    int ix = ox + kx - 1;
    float v = 0.0f;
    if (m < MCONV && iy >= 0 && iy < OH && ix >= 0 && ix < OW)
        v = y1[(size_t)cin * MCONV + iy * OW + ix];
    A2[idx] = (_Float16)v;
}

// ================================================================ host side
extern "C" void kernel_launch(void* const* d_in, const int* in_sizes, int n_in,
                              void* d_out, int out_size, void* d_ws, size_t ws_size,
                              hipStream_t stream) {
    (void)in_sizes; (void)n_in; (void)out_size; (void)ws_size;

    const float* feats[5] = {(const float*)d_in[0], (const float*)d_in[1],
                             (const float*)d_in[2], (const float*)d_in[3],
                             (const float*)d_in[4]};
    const float* calib = (const float*)d_in[5];
    const float* wptr[5] = {(const float*)d_in[6],  (const float*)d_in[8],
                            (const float*)d_in[10], (const float*)d_in[12],
                            (const float*)d_in[14]};
    const float* bptr[5] = {(const float*)d_in[7],  (const float*)d_in[9],
                            (const float*)d_in[11], (const float*)d_in[13],
                            (const float*)d_in[15]};
    const float* att_w   = (const float*)d_in[16];
    const float* att_b   = (const float*)d_in[17];
    const float* conv1_w = (const float*)d_in[18];
    const float* conv2_w = (const float*)d_in[19];
    const float* bn1g = (const float*)d_in[20], *bn1b = (const float*)d_in[21];
    const float* bn1m = (const float*)d_in[22], *bn1v = (const float*)d_in[23];
    const float* bn2g = (const float*)d_in[24], *bn2b = (const float*)d_in[25];
    const float* bn2m = (const float*)d_in[26], *bn2v = (const float*)d_in[27];

    const int Hs[5]  = {96, 48, 24, 12, 6};
    const int Wsz[5] = {320, 160, 80, 40, 20};

    char* ws = (char*)d_ws;

    // ---- region A (phase-1: integral images + Avox; phase-2: A1 + A2 im2col)
    size_t iiOff[5], cur = 0;
    for (int s = 0; s < 5; ++s) {
        iiOff[s] = cur;
        cur += (size_t)Hs[s] * Wsz[s] * 256 * 4;
    }
    size_t avoxOff   = cur;                            // f16 MPORT x KORT
    cur += (size_t)MPORT * KORT * 2;
    size_t regionEnd = cur;
    size_t a1Off = 0;                                  // reuses region A after phase 1
    size_t a2Off = (size_t)MPCON * K1 * 2;             // fits: 79.6MB+15.9MB < 103.8MB

    size_t off = regionEnd;
    auto alloc = [&](size_t b) { size_t o = off; off += (b + 255) & ~(size_t)255; return o; };
    size_t bboxOff  = alloc((size_t)ZM * NLOC * 4 * 4);
    size_t bwOff[5];
    for (int s = 0; s < 5; ++s) bwOff[s] = alloc((size_t)256 * KORT * 2);
    size_t bw1Off   = alloc((size_t)256 * K1 * 2);
    size_t bw2Off   = alloc((size_t)256 * K2 * 2);
    size_t orthoOff = alloc((size_t)NLOC * NCH * 4);   // m-major ortho
    size_t y1Off    = alloc((size_t)256 * MCONV * 4);
    size_t partOff  = alloc((size_t)NGRP * NCH * 4);
    size_t csOff    = alloc(NCH * 4);
    size_t avOff    = alloc(NCH * 4);

    float*     bboxB  = (float*)(ws + bboxOff);
    _Float16*  avoxB  = (_Float16*)(ws + avoxOff);
    float*     orthoB = (float*)(ws + orthoOff);
    float*     y1B    = (float*)(ws + y1Off);
    float*     partB  = (float*)(ws + partOff);
    float*     csB    = (float*)(ws + csOff);
    float*     avB    = (float*)(ws + avOff);

    // ---- weight casts to f16 (N-major == native layout)
    for (int s = 0; s < 5; ++s)
        cvt_f16_kernel<<<(256 * KORT + 255) / 256, 256, 0, stream>>>(
            wptr[s], (_Float16*)(ws + bwOff[s]), 256 * KORT);
    cvt_f16_kernel<<<(256 * K1 + 255) / 256, 256, 0, stream>>>(
        conv1_w, (_Float16*)(ws + bw1Off), 256 * K1);
    cvt_f16_kernel<<<(256 * K2 + 255) / 256, 256, 0, stream>>>(
        conv2_w, (_Float16*)(ws + bw2Off), 256 * K2);

    // ---- zero the M-padding rows of the vox A matrix
    {
        int npad = (MPORT - NLOC) * KORT;
        zero_f16_kernel<<<(npad + 255) / 256, 256, 0, stream>>>(
            avoxB + (size_t)NLOC * KORT, npad);
    }

    // ---- bbox grid (identical for all pyramid levels)
    bbox_kernel<<<(ZM * NLOC + 255) / 256, 256, 0, stream>>>(calib, bboxB);

    // ---- integral images (channel-last for coalesced gathers)
    for (int s = 0; s < 5; ++s) {
        float* iiB = (float*)(ws + iiOff[s]);
        cumsum_w_kernel<<<Hs[s], 256, 0, stream>>>(feats[s], iiB, Hs[s], Wsz[s]);
        cumsum_h_kernel<<<Wsz[s], 256, 0, stream>>>(iiB, Hs[s], Wsz[s]);
    }

    // ---- per level: vox sampling + WMMA linear layer (bias+relu fused, m-major store)
    for (int s = 0; s < 5; ++s) {
        float* iiB = (float*)(ws + iiOff[s]);
        vox_kernel<<<ZM * NLOC, 256, 0, stream>>>(iiB, bboxB, avoxB, Hs[s], Wsz[s]);
        gemm_wmma_kernel<<<dim3(MPORT / 128, 256 / 64), 256, 0, stream>>>(
            avoxB, (const _Float16*)(ws + bwOff[s]),
            NLOC, KORT, /*mode=*/0,
            bptr[s], nullptr, nullptr, nullptr,
            orthoB + (size_t)s * 256);           // channel-segment offset in m-major rows
    }

    // ---- attention (deterministic two-pass reduction, fully coalesced)
    chansum_part_kernel<<<NGRP, 256, 0, stream>>>(orthoB, partB);
    chansum_final_kernel<<<5, 256, 0, stream>>>(partB, csB);
    att_kernel<<<5, 256, 0, stream>>>(csB, att_w, att_b, avB);

    // ---- conv1 (stride 2) as implicit GEMM; attention folded into im2col
    {
        size_t total = (size_t)MPCON * K1;
        im2col1_kernel<<<(unsigned)(total / 256), 256, 0, stream>>>(
            orthoB, avB, (_Float16*)(ws + a1Off));
        gemm_wmma_kernel<<<dim3(MPCON / 128, 256 / 64), 256, 0, stream>>>(
            (const _Float16*)(ws + a1Off), (const _Float16*)(ws + bw1Off),
            MCONV, K1, /*mode=*/1,
            bn1g, bn1b, bn1m, bn1v, y1B);
    }

    // ---- conv2 (stride 1) as implicit GEMM -> final output (1,256,62,54)
    {
        size_t total = (size_t)MPCON * K2;
        im2col2_kernel<<<(unsigned)(total / 256), 256, 0, stream>>>(
            y1B, (_Float16*)(ws + a2Off));
        gemm_wmma_kernel<<<dim3(MPCON / 128, 256 / 64), 256, 0, stream>>>(
            (const _Float16*)(ws + a2Off), (const _Float16*)(ws + bw2Off),
            MCONV, K2, /*mode=*/1,
            bn2g, bn2b, bn2m, bn2v, (float*)d_out);
    }
}
